// GCN_523986010649
// MI455X (gfx1250) — matrix operand order repaired and checked by reading, not compiled, not run on hardware
//
#include <hip/hip_runtime.h>

// ---------------------------------------------------------------------------
// GCN 3-layer forward for MI455X (gfx1250, wave32, WMMA).
//
// Roofline: GEMMs total ~29.5 GFLOP -> negligible on the bf16 WMMA path.
// Edge scatter/gather dominates: ~614MB gather + 153.6M f32 atomics per
// layer, all L2-resident (h/out buffers ~102MB each, 192MB L2).
// GEMM: bf16 WMMA, f32 accumulate, LDS-staged B slab, software-pipelined:
//   - global loads for next B slab + next A tile issued before compute
//   - B operand registers double-buffered across the 16-wide nt unroll
// Aggregation: wave-uniform edge indices scalarized via readfirstlane.
// ---------------------------------------------------------------------------

#define N_NODES 100000
#define N_EDGES 600000
#define IN_CH   64
#define HID     256
#define MTILES  (N_NODES / 16)   // 6250 exactly

typedef __attribute__((ext_vector_type(16))) __bf16        v16bf;
typedef __attribute__((ext_vector_type(8)))  float         v8f;
typedef __attribute__((ext_vector_type(4)))  unsigned int  v4u;

union BfPack {
    v16bf v;
    v4u   q[2];
    unsigned short s[16];
};

__device__ __forceinline__ unsigned short f2bf(float f) {
    unsigned int u = __float_as_uint(f);
    u += 0x7FFFu + ((u >> 16) & 1u);   // round-to-nearest-even
    return (unsigned short)(u >> 16);
}

// --- preprocessing ----------------------------------------------------------

__global__ void cvt_f32_bf16_kernel(const float* __restrict__ x,
                                    unsigned short* __restrict__ a, int n) {
    int i = blockIdx.x * blockDim.x + threadIdx.x;
    if (i < n) a[i] = f2bf(x[i]);
}

__global__ void degree_kernel(const long long* __restrict__ ei,
                              float* __restrict__ deg) {
    int e = blockIdx.x * blockDim.x + threadIdx.x;
    if (e < N_EDGES) atomicAdd(&deg[(int)ei[N_EDGES + e]], 1.0f);
}

__global__ void dinv_kernel(float* __restrict__ deg_dinv) {
    int i = blockIdx.x * blockDim.x + threadIdx.x;
    if (i < N_NODES) {
        float d = deg_dinv[i];
        deg_dinv[i] = (d > 0.0f) ? rsqrtf(fmaxf(d, 1.0f)) : 0.0f;
    }
}

__global__ void norm_kernel(const long long* __restrict__ ei,
                            const float* __restrict__ dinv,
                            float* __restrict__ nrm) {
    int e = blockIdx.x * blockDim.x + threadIdx.x;
    if (e < N_EDGES)
        nrm[e] = dinv[(int)ei[e]] * dinv[(int)ei[N_EDGES + e]];
}

// --- pack W (K x 256, row-major f32) into WMMA B-operand lane layout --------
// B 32x16 bf16 tile, this code's k order (A loaded to match): lane holds col n = lane&15; per-lane 16
// consecutive K values starting at (lane>>4)*16 within the 32-wide k-tile.
// Layout: Bp[kt][nt][lane][16 bf16], so each lane reads contiguous 32B.

__global__ void pack_w_kernel(const float* __restrict__ W,
                              unsigned short* __restrict__ Bp, int K) {
    int tid = blockIdx.x * blockDim.x + threadIdx.x;
    int total = (K / 32) * 16 * 32;     // (kt, nt, lane)
    if (tid >= total) return;
    int lane = tid & 31;
    int nt   = (tid >> 5) & 15;
    int kt   = tid >> 9;
    int n  = nt * 16 + (lane & 15);
    int kb = kt * 32 + (lane >> 4) * 16;
    BfPack p;
#pragma unroll
    for (int i = 0; i < 16; ++i) p.s[i] = f2bf(W[(kb + i) * HID + n]);
    v4u* o = (v4u*)Bp + (size_t)tid * 2;
    o[0] = p.q[0];
    o[1] = p.q[1];
}

// --- GEMM: H[M x 256] = A[M x K](bf16) * W[K x 256] via v_wmma --------------
// 256-thread block = 8 waves; each wave owns one 16-row M-tile and the full
// 256-wide output strip (16 f32 accumulators). Software-pipelined:
// prefetch (slab kt+1, A kt+1) into regs -> 16 wmma on slab kt -> barrier ->
// ds_store prefetched slab -> barrier.

__global__ void __launch_bounds__(256)
gemm_wmma_kernel(const unsigned short* __restrict__ A,
                 const v4u* __restrict__ Bp,
                 float* __restrict__ H, int K) {
    __shared__ v4u ldsB[1024];                    // 16 KB: one 32x256 B slab

    const int lane  = threadIdx.x & 31;
    const int wave  = threadIdx.x >> 5;
    const int mtile = blockIdx.x * 8 + wave;
    const bool valid = (mtile < MTILES);

    int mrow = mtile * 16 + (lane & 15);          // A row for this lane
    if (mrow >= N_NODES) mrow = N_NODES - 1;      // tail waves: safe clamp
    const int khalf  = lane >> 4;
    const int ktiles = K >> 5;

    v8f acc[16];
#pragma unroll
    for (int nt = 0; nt < 16; ++nt) acc[nt] = (v8f)0.0f;

    const unsigned short* arow = A + (size_t)mrow * K;

    // ---- prologue: stage slab 0 in LDS, A tile 0 in regs ----
    v4u pre[4];
#pragma unroll
    for (int j = 0; j < 4; ++j) pre[j] = Bp[threadIdx.x * 4 + j];
#pragma unroll
    for (int j = 0; j < 4; ++j) ldsB[threadIdx.x * 4 + j] = pre[j];

    // A 16x32 bf16 tile, this code's k order (B packed to match): lanes0-15 K in [0,16),
    // lanes16-31 K in [16,32)  -> two 16B chunks per lane.
    BfPack a_cur;
    a_cur.q[0] = *(const v4u*)(arow + khalf * 16);
    a_cur.q[1] = *(const v4u*)(arow + khalf * 16 + 8);
    __syncthreads();

    for (int kt = 0; kt < ktiles; ++kt) {
        const bool have_next = (kt + 1) < ktiles;

        // prefetch next slab + next A into registers (latency hidden
        // under the 16 wmma + 32 ds_loads below)
        BfPack a_next;
        if (have_next) {
            const v4u* src = Bp + (kt + 1) * 1024;
#pragma unroll
            for (int j = 0; j < 4; ++j) pre[j] = src[threadIdx.x * 4 + j];
            const unsigned short* ap = arow + (kt + 1) * 32 + khalf * 16;
            a_next.q[0] = *(const v4u*)(ap);
            a_next.q[1] = *(const v4u*)(ap + 8);
        }

        // compute: B operand registers double-buffered across nt unroll
        BfPack b_cur, b_next;
        b_cur.q[0] = ldsB[lane * 2];
        b_cur.q[1] = ldsB[lane * 2 + 1];
#pragma unroll
        for (int nt = 0; nt < 16; ++nt) {
            if (nt + 1 < 16) {
                const int bi = ((nt + 1) * 32 + lane) * 2;
                b_next.q[0] = ldsB[bi];
                b_next.q[1] = ldsB[bi + 1];
            }
            acc[nt] = __builtin_amdgcn_wmma_f32_16x16x32_bf16(
                false, a_cur.v, false, b_cur.v, (short)0, acc[nt], false, false);
            b_cur = b_next;                       // SSA rename under full unroll
        }

        if (have_next) {
            __syncthreads();                      // all waves done reading slab kt
#pragma unroll
            for (int j = 0; j < 4; ++j) ldsB[threadIdx.x * 4 + j] = pre[j];
            __syncthreads();                      // slab kt+1 visible
            a_cur = a_next;
        }
    }

    if (!valid) return;
    // C/D layout: lane holds col n=lane&15; VGPR r holds row khalf*8 + r.
    const int m0 = mtile * 16 + khalf * 8;
    const int c  = lane & 15;
#pragma unroll
    for (int nt = 0; nt < 16; ++nt) {
        float* o = H + (size_t)m0 * HID + nt * 16 + c;
#pragma unroll
        for (int r = 0; r < 8; ++r) o[(size_t)r * HID] = acc[nt][r];
    }
}

// --- edge scatter: out[dst] += h[src] * norm (64 threads per edge) ----------

__global__ void aggregate_kernel(const long long* __restrict__ ei,
                                 const float* __restrict__ nrm,
                                 const float* __restrict__ H,
                                 float* __restrict__ Out) {
    long long tid = (long long)blockIdx.x * blockDim.x + threadIdx.x;
    int e = (int)(tid >> 6);
    int f = ((int)tid & 63) * 4;
    if (e >= N_EDGES) return;
    // e is wave-uniform (64 threads per edge): scalarize so the index/norm
    // loads become s_loads instead of 32x-redundant vector loads.
    e = __builtin_amdgcn_readfirstlane(e);
    int s = (int)ei[e];
    int d = (int)ei[N_EDGES + e];
    float nr = nrm[e];
    const float4 h = *(const float4*)(H + (size_t)s * HID + f);
    float* o = Out + (size_t)d * HID + f;
    atomicAdd(o + 0, h.x * nr);
    atomicAdd(o + 1, h.y * nr);
    atomicAdd(o + 2, h.z * nr);
    atomicAdd(o + 3, h.w * nr);
}

// --- epilogue: y = elu(agg + b); emit bf16 A for next layer and/or f32 out --

__global__ void bias_elu_kernel(const float* __restrict__ Agg,
                                const float* __restrict__ b,
                                float* __restrict__ outF,
                                unsigned short* __restrict__ outB) {
    int i = blockIdx.x * blockDim.x + threadIdx.x;
    if (i >= N_NODES * HID) return;
    float v = Agg[i] + b[i & (HID - 1)];
    float y = (v > 0.0f) ? v : expm1f(v);
    if (outF) outF[i] = y;
    if (outB) outB[i] = f2bf(y);
}

// ---------------------------------------------------------------------------

extern "C" void kernel_launch(void* const* d_in, const int* in_sizes, int n_in,
                              void* d_out, int out_size, void* d_ws, size_t ws_size,
                              hipStream_t stream) {
    const float*     x  = (const float*)d_in[0];
    const long long* ei = (const long long*)d_in[1];
    const float* W1 = (const float*)d_in[2];
    const float* b1 = (const float*)d_in[3];
    const float* W2 = (const float*)d_in[4];
    const float* b2 = (const float*)d_in[5];
    const float* W3 = (const float*)d_in[6];
    const float* b3 = (const float*)d_in[7];
    float* out = (float*)d_out;

    // workspace carve-up (~157 MB)
    char* ws = (char*)d_ws;
    size_t off = 0;
    auto carve = [&](size_t bytes) -> void* {
        void* p = ws + off;
        off += (bytes + 255) & ~(size_t)255;
        return p;
    };
    unsigned short* Abuf = (unsigned short*)carve((size_t)N_NODES * HID * 2);
    float*          Hbuf = (float*)carve((size_t)N_NODES * HID * 4);
    float*          nrm  = (float*)carve((size_t)N_EDGES * 4);
    float*          dinv = (float*)carve((size_t)N_NODES * 4);
    unsigned short* Bp   = (unsigned short*)carve((size_t)8 * 16 * 32 * 16 * 2);

    const int gemm_blocks = (MTILES + 7) / 8;                 // 782
    const int agg_blocks  = (N_EDGES * 64) / 256;             // 150000 exact
    const int nh_blocks   = (N_NODES * HID + 255) / 256;

    // degree -> dinv -> per-edge norm (shared by all 3 layers)
    hipMemsetAsync(dinv, 0, (size_t)N_NODES * 4, stream);
    degree_kernel<<<(N_EDGES + 255) / 256, 256, 0, stream>>>(ei, dinv);
    dinv_kernel<<<(N_NODES + 255) / 256, 256, 0, stream>>>(dinv);
    norm_kernel<<<(N_EDGES + 255) / 256, 256, 0, stream>>>(ei, dinv, nrm);

    // A0 = bf16(x)
    cvt_f32_bf16_kernel<<<(N_NODES * IN_CH + 255) / 256, 256, 0, stream>>>(
        x, Abuf, N_NODES * IN_CH);

    // ---- layer 1 (K=64) ----
    pack_w_kernel<<<((64 / 32) * 16 * 32 + 255) / 256, 256, 0, stream>>>(W1, Bp, 64);
    gemm_wmma_kernel<<<gemm_blocks, 256, 0, stream>>>(Abuf, (const v4u*)Bp, Hbuf, 64);
    hipMemsetAsync(out, 0, (size_t)N_NODES * HID * 4, stream);
    aggregate_kernel<<<agg_blocks, 256, 0, stream>>>(ei, nrm, Hbuf, out);
    bias_elu_kernel<<<nh_blocks, 256, 0, stream>>>(out, b1, nullptr, Abuf);

    // ---- layer 2 (K=256) ----
    pack_w_kernel<<<((256 / 32) * 16 * 32 + 255) / 256, 256, 0, stream>>>(W2, Bp, 256);
    gemm_wmma_kernel<<<gemm_blocks, 256, 0, stream>>>(Abuf, (const v4u*)Bp, Hbuf, 256);
    hipMemsetAsync(out, 0, (size_t)N_NODES * HID * 4, stream);
    aggregate_kernel<<<agg_blocks, 256, 0, stream>>>(ei, nrm, Hbuf, out);
    bias_elu_kernel<<<nh_blocks, 256, 0, stream>>>(out, b2, nullptr, Abuf);

    // ---- layer 3 (K=256), f32 result in d_out ----
    pack_w_kernel<<<((256 / 32) * 16 * 32 + 255) / 256, 256, 0, stream>>>(W3, Bp, 256);
    gemm_wmma_kernel<<<gemm_blocks, 256, 0, stream>>>(Abuf, (const v4u*)Bp, Hbuf, 256);
    hipMemsetAsync(out, 0, (size_t)N_NODES * HID * 4, stream);
    aggregate_kernel<<<agg_blocks, 256, 0, stream>>>(ei, nrm, Hbuf, out);
    bias_elu_kernel<<<nh_blocks, 256, 0, stream>>>(out, b3, out, nullptr);
}
